// WaveletAttention_30777735643482
// MI455X (gfx1250) — matrix operand order repaired and checked by reading, not compiled, not run on hardware
//
#include <hip/hip_runtime.h>

// ---------------------------------------------------------------------------
// CDNA5 (gfx1250) wavelet-attention. wave32, bf16 WMMA 16x16x32, f32 accum.
// Q/K head-major (B,8,n,64), V dim-major (B,8,64,n): every WMMA fragment load
// is a contiguous b128 per lane. GEMM is double-buffered; attention issues
// K then V loads up-front so waits are partial and V flight covers softmax.
// ---------------------------------------------------------------------------

typedef __bf16 bf16_t;
typedef bf16_t v16bf __attribute__((ext_vector_type(16)));
typedef float  v8f   __attribute__((ext_vector_type(8)));

union FragB { v16bf v; unsigned u[8]; };

__device__ __forceinline__ unsigned pack2(float a, float b) {
    bf16_t x = (bf16_t)a, y = (bf16_t)b;
    return (unsigned)__builtin_bit_cast(unsigned short, x) |
           ((unsigned)__builtin_bit_cast(unsigned short, y) << 16);
}

__device__ __forceinline__ v8f vzero8() {
    v8f r;
#pragma unroll
    for (int i = 0; i < 8; ++i) r[i] = 0.f;
    return r;
}

// ---------------------------------------------------------------------------
// f32 -> bf16 elementwise (8 elements/thread, b128 in, b128 out).
// ---------------------------------------------------------------------------
__global__ void f32_to_bf16_kernel(const float* __restrict__ src,
                                   unsigned short* __restrict__ dst, long n)
{
    long i = ((long)blockIdx.x * blockDim.x + threadIdx.x) * 8;
    if (i >= n) return;
    const float4* s = (const float4*)(src + i);
    float4 a = s[0], b = s[1];
    uint4 o;
    o.x = pack2(a.x, a.y); o.y = pack2(a.z, a.w);
    o.z = pack2(b.x, b.y); o.w = pack2(b.z, b.w);
    *(uint4*)(dst + i) = o;
}

// ---------------------------------------------------------------------------
// W (Kd x Nout, f32 row-major) -> Wt (Nout x Kd, bf16): column-major weights.
// ---------------------------------------------------------------------------
__global__ void w_transpose_bf16(const float* __restrict__ W,
                                 unsigned short* __restrict__ Wt,
                                 int Kd, int Nout)
{
    long gid = (long)blockIdx.x * blockDim.x + threadIdx.x;
    long total = (long)Nout * Kd / 8;
    if (gid >= total) return;
    int kb8 = Kd / 8;
    int col = (int)(gid / kb8);
    int k0  = (int)(gid % kb8) * 8;
    float v[8];
#pragma unroll
    for (int j = 0; j < 8; ++j) v[j] = W[(long)(k0 + j) * Nout + col];
    uint4 o;
    o.x = pack2(v[0], v[1]); o.y = pack2(v[2], v[3]);
    o.z = pack2(v[4], v[5]); o.w = pack2(v[6], v[7]);
    *(uint4*)(Wt + (long)col * Kd + k0) = o;
}

// ---------------------------------------------------------------------------
// GEMM + bias on pre-converted bf16. One wave -> 16x64 tile; 2D grid
// (mtiles, ctiles). Double-buffered: next k-step's 10 b128 loads issued
// before current step's 4 WMMAs execute.
// mode 0: row-major ; 1: head-major (B,8,tokN,64) ; 2: dim-major (B,8,64,tokN)
// ---------------------------------------------------------------------------
__global__ __launch_bounds__(32)
void gemm_bias_wmma(const unsigned short* __restrict__ Abf,   // M x Kd bf16
                    const unsigned short* __restrict__ Wt,    // Nout x Kd bf16
                    const float* __restrict__ bias, float* __restrict__ out,
                    int Kd, int Nout, int mode, int tokN)
{
    const int lane  = threadIdx.x;
    const int col16 = lane & 15;
    const int hi    = lane >> 4;
    const int row0  = blockIdx.x * 16;
    const int c0    = blockIdx.y * 64;

    v8f acc[4];
#pragma unroll
    for (int j = 0; j < 4; ++j) acc[j] = vzero8();

    const uint4* arow = (const uint4*)(Abf + (long)(row0 + col16) * Kd);
    const uint4* wrow[4];
#pragma unroll
    for (int j = 0; j < 4; ++j)
        wrow[j] = (const uint4*)(Wt + (long)(c0 + j * 16 + col16) * Kd);

    // stage 0 loads
    uint4 ca0 = arow[2 * hi], ca1 = arow[2 * hi + 1];
    uint4 cb0[4], cb1[4];
#pragma unroll
    for (int j = 0; j < 4; ++j) { cb0[j] = wrow[j][2 * hi]; cb1[j] = wrow[j][2 * hi + 1]; }

    for (int kk = 0; kk < Kd; kk += 32) {
        const int nx = (kk + 32 < Kd ? kk + 32 : kk) / 8;   // next step (or harmless reload)
        uint4 na0 = arow[nx + 2 * hi], na1 = arow[nx + 2 * hi + 1];
        uint4 nb0[4], nb1[4];
#pragma unroll
        for (int j = 0; j < 4; ++j) { nb0[j] = wrow[j][nx + 2 * hi]; nb1[j] = wrow[j][nx + 2 * hi + 1]; }

        FragB af;
        af.u[0] = ca0.x; af.u[1] = ca0.y; af.u[2] = ca0.z; af.u[3] = ca0.w;
        af.u[4] = ca1.x; af.u[5] = ca1.y; af.u[6] = ca1.z; af.u[7] = ca1.w;
#pragma unroll
        for (int j = 0; j < 4; ++j) {
            FragB bfr;
            bfr.u[0] = cb0[j].x; bfr.u[1] = cb0[j].y; bfr.u[2] = cb0[j].z; bfr.u[3] = cb0[j].w;
            bfr.u[4] = cb1[j].x; bfr.u[5] = cb1[j].y; bfr.u[6] = cb1[j].z; bfr.u[7] = cb1[j].w;
            acc[j] = __builtin_amdgcn_wmma_f32_16x16x32_bf16(
                false, af.v, false, bfr.v, (short)0, acc[j], false, false);
        }
        ca0 = na0; ca1 = na1;
#pragma unroll
        for (int j = 0; j < 4; ++j) { cb0[j] = nb0[j]; cb1[j] = nb1[j]; }
    }

#pragma unroll
    for (int j = 0; j < 4; ++j) {
        int wc = c0 + j * 16 + col16;
        float bv = bias[wc];
#pragma unroll
        for (int r = 0; r < 8; ++r) {
            int row = row0 + r + 8 * hi;
            float val = acc[j][r] + bv;
            if (mode == 0) {
                out[(long)row * Nout + wc] = val;
            } else {
                int bb = row >> 12, t = row & (tokN - 1);       // tokN = 4096
                int c = wc >> 3, hh = wc & 7;
                if (mode == 1)
                    out[((long)(bb * 8 + hh) * tokN + t) * 64 + c] = val;
                else
                    out[((long)(bb * 8 + hh) * 64 + c) * (long)tokN + t] = val;
            }
        }
    }
}

// ---------------------------------------------------------------------------
// Wavelet step, parametric strides. Grid: (nh*64/256, B); nh is a power of 2.
// ---------------------------------------------------------------------------
__global__ void wavelet_kernel(const float* __restrict__ x,
                               const float* __restrict__ ecd,
                               const float* __restrict__ ecs,
                               float* __restrict__ dcoef,
                               float* __restrict__ scoef, int nh,
                               int tsi, int csi, int hsi,
                               int tso, int cso, int hso)
{
    int gid = blockIdx.x * blockDim.x + threadIdx.x;    // < nh*64
    int c = gid & 63;
    int i = gid >> 6;
    int b = blockIdx.y;

    long ibase = (long)b * (1024L * nh) + (long)(2 * i) * tsi + (long)c * csi;
    float xa[16];
#pragma unroll
    for (int h = 0; h < 8; ++h) {
        xa[h]     = x[ibase + (long)h * hsi];
        xa[8 + h] = x[ibase + tsi + (long)h * hsi];
    }
    float d[8], s[8];
#pragma unroll
    for (int hh = 0; hh < 8; ++hh) { d[hh] = 0.f; s[hh] = 0.f; }
    for (int j = 0; j < 16; ++j) {
        float xv = xa[j];
#pragma unroll
        for (int hh = 0; hh < 8; ++hh) {
            d[hh] = fmaf(xv, ecd[j * 8 + hh], d[hh]);
            s[hh] = fmaf(xv, ecs[j * 8 + hh], s[hh]);
        }
    }
    long obase = (long)b * (512L * nh) + (long)i * tso + (long)c * cso;
#pragma unroll
    for (int hh = 0; hh < 8; ++hh) {
        dcoef[obase + (long)hh * hso] = d[hh];
        scoef[obase + (long)hh * hso] = s[hh];
    }
}

// ---------------------------------------------------------------------------
// Flash attention (transposed-score form). Grid: (qtiles, B*8), one wave per
// block. K loads then V loads are issued before any consumption: the
// pre-score wait covers only K (in-order completion), V stays in flight
// across softmax. Next chunk's K rows are prefetched (global_prefetch_b8).
// ---------------------------------------------------------------------------
__global__ __launch_bounds__(32)
void attn_kernel(const float* __restrict__ Qhm, const float* __restrict__ Khm,
                 const float* __restrict__ Vdm, float* __restrict__ Ohm,
                 int n, float scale)
{
    const int lane = threadIdx.x;
    const int col  = lane & 15;
    const int hi   = lane >> 4;

    const int qt = blockIdx.x;
    const int h  = blockIdx.y & 7;
    const int b  = blockIdx.y >> 3;

    const float* Qb = Qhm + (long)(b * 8 + h) * n * 64;
    const float* Kb = Khm + (long)(b * 8 + h) * n * 64;
    const float* Vb = Vdm + (long)(b * 8 + h) * 64 * n;
    float*       Ob = Ohm + (long)(b * 8 + h) * n * 64;

    const float4 z4 = make_float4(0.f, 0.f, 0.f, 0.f);

    // Q^T as two B fragments: lane col = token, K rows = c runs {8hi..+7, 16+8hi..+7}.
    const int  qtok = qt * 16 + col;
    const bool qok  = qtok < n;
    FragB qf[2];
    {
        const float4* qp = (const float4*)(Qb + (long)qtok * 64);
#pragma unroll
        for (int ch = 0; ch < 2; ++ch) {
            int c4 = (ch * 32 + 8 * hi) >> 2;
            float4 a0 = qok ? qp[c4]     : z4;
            float4 a1 = qok ? qp[c4 + 1] : z4;
            float4 a2 = qok ? qp[c4 + 4] : z4;
            float4 a3 = qok ? qp[c4 + 5] : z4;
            qf[ch].u[0] = pack2(a0.x, a0.y); qf[ch].u[1] = pack2(a0.z, a0.w);
            qf[ch].u[2] = pack2(a1.x, a1.y); qf[ch].u[3] = pack2(a1.z, a1.w);
            qf[ch].u[4] = pack2(a2.x, a2.y); qf[ch].u[5] = pack2(a2.z, a2.w);
            qf[ch].u[6] = pack2(a3.x, a3.y); qf[ch].u[7] = pack2(a3.z, a3.w);
        }
    }

    v8f o[4];
#pragma unroll
    for (int t = 0; t < 4; ++t) o[t] = vzero8();
    float mprev = -3.0e38f, lsum = 0.f;

    for (int kb = 0; kb < n; kb += 32) {
        // ---- issue K loads (consumed first) ----
        float4 kr[2][2][4];
#pragma unroll
        for (int half = 0; half < 2; ++half) {
            int  ktok = kb + half * 16 + col;
            bool kok  = ktok < n;
            const float4* kp = (const float4*)(Kb + (long)ktok * 64);
#pragma unroll
            for (int ch = 0; ch < 2; ++ch) {
                int r1 = (ch * 32 + 8 * hi) >> 2;
                int r2 = (ch * 32 + 16 + 8 * hi) >> 2;
                kr[half][ch][0] = kok ? kp[r1]     : z4;
                kr[half][ch][1] = kok ? kp[r1 + 1] : z4;
                kr[half][ch][2] = kok ? kp[r2]     : z4;
                kr[half][ch][3] = kok ? kp[r2 + 1] : z4;
            }
        }
        // ---- issue V loads (consumed after softmax; stay in flight) ----
        float4 vr[4][4];
        {
            const bool v1ok = (kb + 16 * hi) < n;       // 8-aligned runs, n%8==0
            const bool v2ok = (kb + 16 * hi + 8) < n;
            const int  r1 = (kb + 16 * hi) >> 2;
            const int  r2 = (kb + 16 * hi + 8) >> 2;
#pragma unroll
            for (int dt = 0; dt < 4; ++dt) {
                const float4* vp = (const float4*)(Vb + (long)(dt * 16 + col) * n);
                vr[dt][0] = v1ok ? vp[r1]     : z4;
                vr[dt][1] = v1ok ? vp[r1 + 1] : z4;
                vr[dt][2] = v2ok ? vp[r2]     : z4;
                vr[dt][3] = v2ok ? vp[r2 + 1] : z4;
            }
        }
        // ---- prefetch next chunk's K rows ----
        if (kb + 32 < n) {
            __builtin_prefetch(Kb + (long)(kb + 32 + col) * 64 + 8 * hi, 0, 3);
            __builtin_prefetch(Kb + (long)(kb + 48 + col) * 64 + 8 * hi, 0, 3);
        }

        // ---- scores: two 16x16 S^T tiles ----
        v8f s0 = vzero8(), s1 = vzero8();
#pragma unroll
        for (int half = 0; half < 2; ++half) {
            v8f accS = vzero8();
#pragma unroll
            for (int ch = 0; ch < 2; ++ch) {
                FragB kf;
                kf.u[0] = pack2(kr[half][ch][0].x, kr[half][ch][0].y);
                kf.u[1] = pack2(kr[half][ch][0].z, kr[half][ch][0].w);
                kf.u[2] = pack2(kr[half][ch][1].x, kr[half][ch][1].y);
                kf.u[3] = pack2(kr[half][ch][1].z, kr[half][ch][1].w);
                kf.u[4] = pack2(kr[half][ch][2].x, kr[half][ch][2].y);
                kf.u[5] = pack2(kr[half][ch][2].z, kr[half][ch][2].w);
                kf.u[6] = pack2(kr[half][ch][3].x, kr[half][ch][3].y);
                kf.u[7] = pack2(kr[half][ch][3].z, kr[half][ch][3].w);
                accS = __builtin_amdgcn_wmma_f32_16x16x32_bf16(
                    false, kf.v, false, qf[ch].v, (short)0, accS, false, false);
            }
            if (half == 0) s0 = accS; else s1 = accS;
        }

        // ---- scale + mask (C tile: key = kb + r + 8hi) ----
#pragma unroll
        for (int r = 0; r < 8; ++r) {
            s0[r] = (kb + r + 8 * hi >= n)      ? -3.0e38f : s0[r] * scale;
            s1[r] = (kb + 16 + r + 8 * hi >= n) ? -3.0e38f : s1[r] * scale;
        }

        // ---- online softmax (per query column) ----
        float cm = s0[0];
#pragma unroll
        for (int r = 1; r < 8; ++r) cm = fmaxf(cm, s0[r]);
#pragma unroll
        for (int r = 0; r < 8; ++r) cm = fmaxf(cm, s1[r]);
        cm = fmaxf(cm, __shfl_xor(cm, 16));
        float mnew = fmaxf(mprev, cm);
        float corr = __expf(mprev - mnew);
        lsum *= corr;
#pragma unroll
        for (int t = 0; t < 4; ++t)
#pragma unroll
            for (int r = 0; r < 8; ++r) o[t][r] *= corr;

        float psum = 0.f;
        v8f p0, p1;
#pragma unroll
        for (int r = 0; r < 8; ++r) { p0[r] = __expf(s0[r] - mnew); psum += p0[r]; }
#pragma unroll
        for (int r = 0; r < 8; ++r) { p1[r] = __expf(s1[r] - mnew); psum += p1[r]; }
        psum += __shfl_xor(psum, 16);
        lsum += psum;
        mprev = mnew;

        // ---- repack P^T into a 32x16 B fragment (half-wave swaps) ----
        FragB pf;
#pragma unroll
        for (int j = 0; j < 4; ++j) {
            float a0 = p0[2 * j], a1 = p0[2 * j + 1];
            float b0 = p1[2 * j], b1 = p1[2 * j + 1];
            float t0 = __shfl_xor(b0, 16), t1 = __shfl_xor(b1, 16);
            float u0 = __shfl_xor(a0, 16), u1 = __shfl_xor(a1, 16);
            pf.u[j]     = hi ? pack2(t0, t1) : pack2(a0, a1);
            pf.u[4 + j] = hi ? pack2(b0, b1) : pack2(u0, u1);
        }

        // ---- O^T += V^T(16dims x 32keys) * P^T ----
#pragma unroll
        for (int dt = 0; dt < 4; ++dt) {
            FragB vf;
            vf.u[0] = pack2(vr[dt][0].x, vr[dt][0].y);
            vf.u[1] = pack2(vr[dt][0].z, vr[dt][0].w);
            vf.u[2] = pack2(vr[dt][1].x, vr[dt][1].y);
            vf.u[3] = pack2(vr[dt][1].z, vr[dt][1].w);
            vf.u[4] = pack2(vr[dt][2].x, vr[dt][2].y);
            vf.u[5] = pack2(vr[dt][2].z, vr[dt][2].w);
            vf.u[6] = pack2(vr[dt][3].x, vr[dt][3].y);
            vf.u[7] = pack2(vr[dt][3].z, vr[dt][3].w);
            o[dt] = __builtin_amdgcn_wmma_f32_16x16x32_bf16(
                false, vf.v, false, pf.v, (short)0, o[dt], false, false);
        }
    }

    const float inv = 1.f / lsum;
    if (qok) {
        float* orow = Ob + (long)qtok * 64;
#pragma unroll
        for (int dt = 0; dt < 4; ++dt) {
            int d0 = dt * 16 + 8 * hi;
            *(float4*)(orow + d0) =
                make_float4(o[dt][0] * inv, o[dt][1] * inv, o[dt][2] * inv, o[dt][3] * inv);
            *(float4*)(orow + d0 + 4) =
                make_float4(o[dt][4] * inv, o[dt][5] * inv, o[dt][6] * inv, o[dt][7] * inv);
        }
    }
}

// ---------------------------------------------------------------------------
// Reconstruction. Grid: (n*64/256, B); n is a power of 2.
// v strides parametric; Us/Ud head-major; output token-major (B,2n,64,8).
// ---------------------------------------------------------------------------
__global__ void recon_kernel(const float* __restrict__ v, int vts, int vcs, int vhs,
                             const float* __restrict__ us,
                             const float* __restrict__ ud,
                             const float* __restrict__ rce,
                             const float* __restrict__ rco,
                             float* __restrict__ out, int n)
{
    int gid = blockIdx.x * blockDim.x + threadIdx.x;    // < n*64
    int c = gid & 63;
    int i = gid >> 6;
    int b = blockIdx.y;

    long vb = (long)b * (512L * n) + (long)i * vts + (long)c * vcs;
    long ub = ((long)(b * 8) * n + i) * 64 + c;         // head-major, +k*64n
    float x[16];
#pragma unroll
    for (int j = 0; j < 8; ++j) {
        x[j]     = v[vb + (long)j * vhs] + us[ub + (long)j * 64 * n];
        x[8 + j] = ud[ub + (long)j * 64 * n];
    }
    float e[8], od[8];
#pragma unroll
    for (int hh = 0; hh < 8; ++hh) { e[hh] = 0.f; od[hh] = 0.f; }
    for (int j = 0; j < 16; ++j) {
        float xv = x[j];
#pragma unroll
        for (int hh = 0; hh < 8; ++hh) {
            e[hh]  = fmaf(xv, rce[j * 8 + hh], e[hh]);
            od[hh] = fmaf(xv, rco[j * 8 + hh], od[hh]);
        }
    }
    long ob = ((long)(b * (2L * n) + 2L * i) * 64 + c) * 8;
#pragma unroll
    for (int hh = 0; hh < 8; ++hh) { out[ob + hh] = e[hh]; out[ob + 512 + hh] = od[hh]; }
}

// ---------------------------------------------------------------------------
// Host orchestration (graph-capture safe).
// ---------------------------------------------------------------------------
extern "C" void kernel_launch(void* const* d_in, const int* in_sizes, int n_in,
                              void* d_out, int out_size, void* d_ws, size_t ws_size,
                              hipStream_t stream)
{
    (void)in_sizes; (void)n_in; (void)out_size; (void)ws_size;

    const float* q    = (const float*)d_in[0];
    const float* k    = (const float*)d_in[1];
    const float* v    = (const float*)d_in[2];
    const float* ec_s = (const float*)d_in[3];
    const float* ec_d = (const float*)d_in[4];
    const float* rc_e = (const float*)d_in[5];
    const float* rc_o = (const float*)d_in[6];
    const float* Wq   = (const float*)d_in[7];
    const float* bq   = (const float*)d_in[8];
    const float* Wk   = (const float*)d_in[9];
    const float* bk   = (const float*)d_in[10];
    const float* Wv   = (const float*)d_in[11];
    const float* bv   = (const float*)d_in[12];
    const float* Wo   = (const float*)d_in[13];
    const float* bo   = (const float*)d_in[14];

    const long B = 2, N = 4096, CH = 512;
    const long MK = B * N * CH;                      // 4,194,304 elements
    float* ws = (float*)d_ws;
    long off = 0;
    auto alloc = [&](long nf) { float* p = ws + off; off += nf; return p; };

    float* Qhm0 = alloc(MK);
    float* Khm0 = alloc(MK);
    float* Vdm0 = alloc(MK);
    const long HALF = B * 2048 * CH;
    float* sq[2] = { alloc(HALF), alloc(HALF) };
    float* sk[2] = { alloc(HALF), alloc(HALF) };
    float* sv[2] = { alloc(HALF), alloc(HALF) };
    float* dq = alloc(HALF); float* dk = alloc(HALF); float* dv = alloc(HALF);

    long usOff[9], usTot = 0;
    for (int i = 0; i < 9; ++i) { usOff[i] = usTot; usTot += B * (2048L >> i) * CH; }
    float* Us = alloc(usTot);
    float* Ud = alloc(usTot);

    unsigned short* Abf = (unsigned short*)alloc(MK / 2);
    unsigned short* Wt  = (unsigned short*)alloc(CH * CH / 2);

    const dim3 wave(32), tb(256);
    const dim3 gemmGrid(8192 / 16, 512 / 64);
    const int  cvtGrid = (int)(MK / 8 / 256);
    const int  wtGrid  = (int)(CH * CH / 8 / 256);

    auto do_gemm = [&](const float* A, const float* W, const float* bias,
                       float* out, int mode) {
        f32_to_bf16_kernel<<<dim3(cvtGrid), tb, 0, stream>>>(A, Abf, MK);
        w_transpose_bf16<<<dim3(wtGrid), tb, 0, stream>>>(W, Wt, 512, 512);
        gemm_bias_wmma<<<gemmGrid, wave, 0, stream>>>(
            Abf, Wt, bias, out, 512, 512, mode, 4096);
    };

    // projections: Q/K head-major, V dim-major
    do_gemm(q, Wq, bq, Qhm0, 1);
    do_gemm(k, Wk, bk, Khm0, 1);
    do_gemm(v, Wv, bv, Vdm0, 2);

    const float scale = 0.125f;                      // 1/sqrt(E)/T
    const float* curQ = Qhm0; const float* curK = Khm0; const float* curV = Vdm0;

    for (int i = 0; i < 9; ++i) {
        int n = 2048 >> i, nin = n * 2;
        dim3 wgrid((unsigned)(n * 64 / 256), (unsigned)B);
        if (n * 64 < 256) wgrid = dim3(1, (unsigned)B);          // n >= 8 -> never hit
        // head-major strides (Q, K)
        wavelet_kernel<<<wgrid, tb, 0, stream>>>(
            curQ, ec_d, ec_s, dq, sq[i & 1], n, 64, 1, nin * 64, 64, 1, n * 64);
        wavelet_kernel<<<wgrid, tb, 0, stream>>>(
            curK, ec_d, ec_s, dk, sk[i & 1], n, 64, 1, nin * 64, 64, 1, n * 64);
        // dim-major strides (V)
        wavelet_kernel<<<wgrid, tb, 0, stream>>>(
            curV, ec_d, ec_s, dv, sv[i & 1], n, 1, nin, 64 * nin, 1, n, 64 * n);
        curQ = sq[i & 1]; curK = sk[i & 1]; curV = sv[i & 1];

        dim3 agrid((unsigned)((n + 15) / 16), (unsigned)(B * 8));
        attn_kernel<<<agrid, wave, 0, stream>>>(dq, dk, dv, Ud + usOff[i], n, scale);
        attn_kernel<<<agrid, wave, 0, stream>>>(curQ, curK, curV, Us + usOff[i], n, scale);
    }

    // reconstruction (level-0 buffers dead; reuse as token-major ping-pong)
    const float* vcur = sv[0];                       // final scale v, dim-major, n=8
    float* rb[2] = { Qhm0, Khm0 };
    for (int step = 0; step < 9; ++step) {
        int i = 8 - step;
        int n = 2048 >> i;
        dim3 rgrid((unsigned)(n * 64 / 256), (unsigned)B);
        int vts, vcs, vhs;
        if (step == 0) { vts = 1;   vcs = n; vhs = 64 * n; }   // dim-major
        else           { vts = 512; vcs = 8; vhs = 1;      }   // token-major
        recon_kernel<<<rgrid, tb, 0, stream>>>(
            vcur, vts, vcs, vhs, Us + usOff[i], Ud + usOff[i],
            rc_e, rc_o, rb[step & 1], n);
        vcur = rb[step & 1];
    }

    // output projection -> d_out (B, N, 512) f32, row-major
    do_gemm(vcur, Wo, bo, (float*)d_out, 0);
}
